// BaselineGraphSAGE_66168266162538
// MI455X (gfx1250) — compile-verified, matrix operand order checked by reading, not run on hardware
//
#include <hip/hip_runtime.h>

// ---------------------------------------------------------------------------
// GraphSAGE (2x SAGEConv mean-aggr + Linear) for MI455X / gfx1250, wave32.
//  - Aggregation: edge-parallel f32 atomics; x/h matrices are 12.8-25.6 MB,
//    fully resident in the 192 MB L2, so random gathers/atomics stay on-chip.
//  - Dense transforms: v_wmma_f32_16x16x32_f16 (f16 in, f32 accumulate).
//  - Weights pre-packed ONCE per launch into WMMA B-fragment lane layout
//    (f16), so each GEMM wave loads a fragment as one contiguous 32B v16h.
// ---------------------------------------------------------------------------

typedef __attribute__((ext_vector_type(16))) _Float16 v16h;
typedef __attribute__((ext_vector_type(8)))  float    v8f;

#define HID 64

// ISA 16-bit A/B 16x32 fragment layout (wave32): lane&15 = matrix row (A) /
// column (B = W^T); half j of the fragment holds
//   K = (j&7) + ((j>>3)<<4) + ((lane&16) ? 8 : 0)   (+ 32*kc for K chunks)
__device__ __forceinline__ v16h load_frag_f32(const float* __restrict__ p,
                                              int hi8, float scale) {
  v16h v;
#pragma unroll
  for (int j = 0; j < 16; ++j) {
    int k = (j & 7) + ((j >> 3) << 4) + hi8;
    v[j] = (_Float16)(p[k] * scale);
  }
  return v;
}

// ------------- one-time weight pre-pack into fragment layout ---------------
// pack[( (n*(K/32)+kc)*2 + hi ) * 16 + j] = f16( W[n, (j&7)+((j>>3)<<4)+hi*8+kc*32] )
template <int K>
__global__ void pack_weight(const float* __restrict__ w,
                            _Float16* __restrict__ pack) {
  int p = blockIdx.x * blockDim.x + threadIdx.x;
  if (p >= 64 * K) return;
  int j  = p & 15;
  int f  = p >> 4;
  int hi = f & 1;
  int t  = f >> 1;
  int kc = t % (K / 32);
  int n  = t / (K / 32);
  int k  = (j & 7) + ((j >> 3) << 4) + hi * 8 + kc * 32;
  pack[p] = (_Float16)w[n * K + k];
}

// ---------------- edge scatter kernels (one wave per edge) -----------------

__global__ void scatter32_count(const long long* __restrict__ src,
                                const long long* __restrict__ dst,
                                const float* __restrict__ x,
                                float* __restrict__ agg,
                                float* __restrict__ cnt,
                                int n_edges) {
  int gid  = blockIdx.x * blockDim.x + threadIdx.x;
  int e    = gid >> 5;
  int lane = gid & 31;
  if (e >= n_edges) return;                 // wave-uniform exit
  int s = (int)src[e];
  int d = (int)dst[e];
  atomicAdd(&agg[d * 32 + lane], x[s * 32 + lane]);   // one 128B line each
  if (lane == 0) atomicAdd(&cnt[d], 1.0f);
}

__global__ void scatter64(const long long* __restrict__ src,
                          const long long* __restrict__ dst,
                          const float* __restrict__ h,
                          float* __restrict__ agg,
                          int n_edges) {
  int gid  = blockIdx.x * blockDim.x + threadIdx.x;
  int e    = gid >> 5;
  int lane = gid & 31;
  if (e >= n_edges) return;
  int s = (int)src[e];
  int d = (int)dst[e];
  atomicAdd(&agg[d * 64 + lane],      h[s * 64 + lane]);
  atomicAdd(&agg[d * 64 + 32 + lane], h[s * 64 + 32 + lane]);
}

// --------------- fused SAGEConv linear: relu(mean@Wl^T + b + h@Wr^T) -------
// One wave computes a 16-row tile of the [N, 64] output: 4 column tiles,
// (K/32)*2 WMMAs each; bias preloaded into the f32 accumulator; fused ReLU.
template <int K>
__global__ __launch_bounds__(256)
void sage_layer(const float* __restrict__ agg,
                const float* __restrict__ cnt,
                const float* __restrict__ hin,
                const _Float16* __restrict__ pw_l,   // pre-packed fragments
                const float* __restrict__ b_l,
                const _Float16* __restrict__ pw_r,   // pre-packed fragments
                float* __restrict__ out,
                int n_nodes) {
  int gid  = blockIdx.x * blockDim.x + threadIdx.x;
  int wave = gid >> 5;
  int lane = gid & 31;
  int row0 = wave * 16;
  if (row0 >= n_nodes) return;              // n_nodes % 16 == 0 (100000/16)

  int m   = lane & 15;
  int hi  = lane >> 4;                      // fragment half selector
  int hi8 = hi * 8;
  int row = row0 + m;

  float inv = 1.0f / fmaxf(cnt[row], 1.0f);
  const float* aggrow = agg + (size_t)row * K;
  const float* hrow   = hin + (size_t)row * K;

  // A-fragments shared across all 4 output column tiles
  v16h aMean[K / 32], aSelf[K / 32];
#pragma unroll
  for (int kc = 0; kc < K / 32; ++kc) {
    aMean[kc] = load_frag_f32(aggrow + kc * 32, hi8, inv);   // mean = agg/cnt
    aSelf[kc] = load_frag_f32(hrow   + kc * 32, hi8, 1.0f);
  }

  int m8 = hi * 8;   // C/D: VGPR r holds M = r + m8, N = lane&15

#pragma unroll
  for (int ct = 0; ct < 4; ++ct) {
    int n = ct * 16 + m;                    // output column (and W row)
    float bias = b_l[n];
    v8f c;
#pragma unroll
    for (int r = 0; r < 8; ++r) c[r] = bias;

#pragma unroll
    for (int kc = 0; kc < K / 32; ++kc) {
      // single 32B contiguous v16h load per B-fragment (pre-packed, f16)
      size_t foff = ((size_t)(n * (K / 32) + kc) * 2 + hi) * 16;
      v16h bL = *(const v16h*)(pw_l + foff);
      c = __builtin_amdgcn_wmma_f32_16x16x32_f16(false, aMean[kc], false, bL,
                                                 (short)0, c, false, false);
      v16h bR = *(const v16h*)(pw_r + foff);
      c = __builtin_amdgcn_wmma_f32_16x16x32_f16(false, aSelf[kc], false, bR,
                                                 (short)0, c, false, false);
    }

#pragma unroll
    for (int r = 0; r < 8; ++r)
      out[(size_t)(row0 + r + m8) * HID + n] = fmaxf(c[r], 0.0f);  // fused ReLU
  }
}

// ------------------------- final classifier [N,64]@[64] --------------------
__global__ void classifier(const float* __restrict__ h,
                           const float* __restrict__ w,
                           const float* __restrict__ b,
                           float* __restrict__ out,
                           int n_nodes) {
  int gid  = blockIdx.x * blockDim.x + threadIdx.x;
  int node = gid >> 5;
  int lane = gid & 31;
  if (node >= n_nodes) return;
  float acc = h[(size_t)node * 64 + lane]      * w[lane] +
              h[(size_t)node * 64 + 32 + lane] * w[32 + lane];
#pragma unroll
  for (int off = 16; off > 0; off >>= 1) acc += __shfl_xor(acc, off, 32);
  if (lane == 0) out[node] = acc + b[0];
}

// ---------------------------------------------------------------------------

extern "C" void kernel_launch(void* const* d_in, const int* in_sizes, int n_in,
                              void* d_out, int out_size, void* d_ws, size_t ws_size,
                              hipStream_t stream) {
  const float*     x     = (const float*)d_in[0];
  const long long* ei    = (const long long*)d_in[1];   // jnp.int64 edge_index
  const float*     w1_l  = (const float*)d_in[2];
  const float*     b1_l  = (const float*)d_in[3];
  const float*     w1_r  = (const float*)d_in[4];
  const float*     w2_l  = (const float*)d_in[5];
  const float*     b2_l  = (const float*)d_in[6];
  const float*     w2_r  = (const float*)d_in[7];
  const float*     w_cls = (const float*)d_in[8];
  const float*     b_cls = (const float*)d_in[9];
  float*           out   = (float*)d_out;

  const int N = in_sizes[0] / 32;   // 100000
  const int E = in_sizes[1] / 2;    // 3200000
  const long long* src = ei;        // edge_index[0]
  const long long* dst = ei + E;    // edge_index[1]

  // workspace layout (floats): [cnt N | agg1 32N | agg2 64N | h1 64N | h2 64N]
  // then packed f16 weights (32B-aligned: 225N*4 bytes is a multiple of 32).
  float* ws   = (float*)d_ws;
  float* cnt  = ws;
  float* agg1 = ws + (size_t)N;
  float* agg2 = ws + (size_t)N * 33;
  float* h1   = ws + (size_t)N * 97;
  float* h2   = ws + (size_t)N * 161;
  _Float16* pk   = (_Float16*)(ws + (size_t)N * 225);
  _Float16* p1_l = pk;                    // 64*32 halves
  _Float16* p1_r = pk + 64 * 32;          // 64*32 halves
  _Float16* p2_l = pk + 2 * 64 * 32;      // 64*64 halves
  _Float16* p2_r = pk + 2 * 64 * 32 + 64 * 64;

  // zero cnt+agg1+agg2 in one capturable memset
  hipMemsetAsync(ws, 0, (size_t)N * 97 * sizeof(float), stream);

  // one-time weight pre-pack into WMMA fragment layout (f16)
  pack_weight<32><<<(64 * 32 + 255) / 256, 256, 0, stream>>>(w1_l, p1_l);
  pack_weight<32><<<(64 * 32 + 255) / 256, 256, 0, stream>>>(w1_r, p1_r);
  pack_weight<64><<<(64 * 64 + 255) / 256, 256, 0, stream>>>(w2_l, p2_l);
  pack_weight<64><<<(64 * 64 + 255) / 256, 256, 0, stream>>>(w2_r, p2_r);

  const int edgeBlocks = (int)(((long long)E * 32 + 255) / 256);
  scatter32_count<<<edgeBlocks, 256, 0, stream>>>(src, dst, x, agg1, cnt, E);

  const int tiles      = (N + 15) / 16;     // one wave per 16-row tile
  const int gemmBlocks = (tiles + 7) / 8;   // 8 waves / 256-thread block
  sage_layer<32><<<gemmBlocks, 256, 0, stream>>>(agg1, cnt, x,  p1_l, b1_l, p1_r, h1, N);

  scatter64<<<edgeBlocks, 256, 0, stream>>>(src, dst, h1, agg2, E);
  sage_layer<64><<<gemmBlocks, 256, 0, stream>>>(agg2, cnt, h1, p2_l, b2_l, p2_r, h2, N);

  const int clsBlocks = (int)(((long long)N * 32 + 255) / 256);
  classifier<<<clsBlocks, 256, 0, stream>>>(h2, w_cls, b_cls, out, N);
}
